// TransformerConditionalDecoderLayer_71683004170503
// MI455X (gfx1250) — compile-verified, matrix operand order checked by reading, not run on hardware
//
#include <hip/hip_runtime.h>
#include <hip/hip_bf16.h>

// ---------------------------------------------------------------------------
// Conditional-DETR decoder layer for MI455X (gfx1250, wave32, WMMA).
// D=768 H=6 HD=128 DFF=2048 L=512 S=1024 B=8. All matmuls run on
// v_wmma_f32_16x16x32_bf16 with f32 accumulation; softmax/LN/residual in f32.
// GEMM: 4 waves / block, double-buffered LDS B-tile, 32x64 C-tile per wave.
// ---------------------------------------------------------------------------

#define Dm   768
#define Hh   6
#define HDm  128
#define DFFm 2048
#define Lm   512
#define Sm   1024
#define Bm   8
#define NT   (Lm * Bm)   // 4096 target tokens
#define NS   (Sm * Bm)   // 8192 memory tokens
#define SCALE 0.0625f    // 1/sqrt(2*HD) = 1/16

typedef __bf16 bf16;
typedef __bf16 v16bf __attribute__((ext_vector_type(16)));
typedef float  v8f   __attribute__((ext_vector_type(8)));

// ---------------------------------------------------------------------------
// Generic strided / batched GEMM:  C[m,n] = sum_k A[m,k] * Bw[n,k]  (+ bias[n])
// Block = 128 threads (4 waves). Block tile = 128(M) x 64(N); each wave owns a
// 32x64 tile (2 A-fragments x 4 B-fragments -> 8 WMMAs per 32-wide K-step).
// The 64x32 B tile is double-buffered through LDS (2 x 4KB): the next tile's
// global load is issued before the compute phase so its latency hides under
// the WMMAs, and a single workgroup barrier per K-step both retires reads of
// the live buffer and publishes the staged one.
// Batch base offsets decompose as (bz/batchH)*X0 + (bz%batchH)*X1.
// ---------------------------------------------------------------------------
__global__ __launch_bounds__(128)
void k_gemm(const bf16* __restrict__ A,  long lda, long aB0, long aB1,
            const bf16* __restrict__ Bw, long ldb, long bB0, long bB1,
            const float* __restrict__ bias,
            float* __restrict__ C, long ldc, long cB0, long cB1,
            int batchH, int K)
{
    __shared__ bf16 smem[2][64 * 32];       // B tile: [n][k] row-major, x2

    const int tid  = threadIdx.x;
    const int lane = tid & 31;
    const int wave = tid >> 5;
    const int m0 = blockIdx.x * 128;
    const int n0 = blockIdx.y * 64;
    const int bz = blockIdx.z;
    const int bq = bz / batchH, br = bz % batchH;
    A  += (long)bq * aB0 + (long)br * aB1;
    Bw += (long)bq * bB0 + (long)br * bB1;
    C  += (long)bq * cB0 + (long)br * cB1;

    // --- staging assignment: thread t loads 16 bf16 of B row (n0 + t/2) ---
    const int  sn   = tid >> 1;             // 0..63
    const long skh  = (tid & 1) * 16;       // k-half within 32-wide tile
    const int  sidx = sn * 32 + (int)skh;
    const bf16* pstage = Bw + (long)(n0 + sn) * ldb + skh;

    // --- per-lane fragment addressing ---
    const int  mrow  = lane & 15;
    const long khalf = (lane >> 4) * 16;    // upper half-wave takes K+16..31
    const int  mw    = m0 + wave * 32;      // this wave's 32-row strip
    const bf16* pa0 = A + (long)(mw      + mrow) * lda + khalf;
    const bf16* pa1 = A + (long)(mw + 16 + mrow) * lda + khalf;
    const int  ridx = (lane & 15) * 32 + (int)khalf;

    v8f acc[8] = {};                        // [mi*4 + nt]

    // prologue: stage first B tile into buffer 0
    {
        v16bf b0 = *(const v16bf*)pstage;
        *(v16bf*)&smem[0][sidx] = b0;
    }
    __syncthreads();

    int buf = 0;
    for (int k0 = 0; k0 < K; k0 += 32) {
        const bool have_next = (k0 + 32 < K);

        // issue next B tile's global load now: latency hides under compute
        v16bf bnext = {};
        if (have_next) {
            bnext = *(const v16bf*)(pstage + k0 + 32);
            __builtin_prefetch(pa0 + k0 + 32, 0, 3);    // global_prefetch_b8
        }

        // compute on the live buffer
        const bf16* sb = &smem[buf][ridx];
        v16bf a0 = *(const v16bf*)(pa0 + k0);
        v16bf a1 = *(const v16bf*)(pa1 + k0);
#pragma unroll
        for (int nt = 0; nt < 4; ++nt) {
            v16bf b = *(const v16bf*)(sb + nt * 16 * 32);   // ds_load
            acc[nt]     = __builtin_amdgcn_wmma_f32_16x16x32_bf16(false, a0, false, b, (short)0, acc[nt],     false, false);
            acc[4 + nt] = __builtin_amdgcn_wmma_f32_16x16x32_bf16(false, a1, false, b, (short)0, acc[4 + nt], false, false);
        }

        // publish next tile into the alternate buffer; one barrier does both:
        // retire reads of smem[buf] and make smem[buf^1] visible
        if (have_next)
            *(v16bf*)&smem[buf ^ 1][sidx] = bnext;
        __syncthreads();
        buf ^= 1;
    }

    // C/D layout (ISA 7.12.2): lanes 0-15 rows r, lanes 16-31 rows 8+r.
    const int ncol  = lane & 15;
    const int rbase = (lane >> 4) * 8;
#pragma unroll
    for (int mi = 0; mi < 2; ++mi) {
#pragma unroll
        for (int nt = 0; nt < 4; ++nt) {
            const int   n  = n0 + nt * 16 + ncol;
            const float bv = bias ? bias[n] : 0.0f;
#pragma unroll
            for (int r = 0; r < 8; ++r)
                C[(long)(mw + mi * 16 + rbase + r) * ldc + n] = acc[mi * 4 + nt][r] + bv;
        }
    }
}

// --------------------------- elementwise kernels ---------------------------

__global__ void k_cvt(const float* __restrict__ x, bf16* __restrict__ y, long n) {
    long i = (long)blockIdx.x * blockDim.x + threadIdx.x;
    if (i < n) y[i] = (bf16)x[i];
}

__global__ void k_addcvt(const float* __restrict__ a, const float* __restrict__ b,
                         bf16* __restrict__ y, long n) {
    long i = (long)blockIdx.x * blockDim.x + threadIdx.x;
    if (i < n) y[i] = (bf16)(a[i] + b[i]);
}

__global__ void k_relu(const float* __restrict__ x, bf16* __restrict__ y, long n) {
    long i = (long)blockIdx.x * blockDim.x + threadIdx.x;
    if (i < n) y[i] = (bf16)fmaxf(x[i], 0.0f);
}

// per-head concat of content/pos projections -> fused (tok, H, 2*HD) bf16
__global__ void k_fuse(const float* __restrict__ c, const float* __restrict__ p,
                       bf16* __restrict__ out, float scale, long n) {
    long i = (long)blockIdx.x * blockDim.x + threadIdx.x;
    if (i >= n) return;
    long tok = i / (2 * Dm);
    int  r   = (int)(i % (2 * Dm));
    int  h   = r >> 8;              // / 256
    int  j   = r & 255;
    const float* src = (j < HDm) ? c : p;
    out[i] = (bf16)(src[tok * Dm + h * HDm + (j & (HDm - 1))] * scale);
}

// V (tok=s*B+b, D) f32 -> Vt (b,h,e,s) bf16 so the AV GEMM has contiguous K
__global__ void k_vt(const float* __restrict__ v, bf16* __restrict__ out,
                     int Slen, long n) {
    long i = (long)blockIdx.x * blockDim.x + threadIdx.x;
    if (i >= n) return;
    long s  = i % Slen;
    long r  = i / Slen;
    int  e  = (int)(r % HDm);
    int  bh = (int)(r / HDm);
    int  b  = bh / Hh, h = bh % Hh;
    out[i] = (bf16)v[(s * Bm + b) * Dm + h * HDm + e];
}

// row softmax over Slen, one wave per (b,h,l) row; writes bf16 probs
__global__ __launch_bounds__(32)
void k_softmax(const float* __restrict__ scores, bf16* __restrict__ probs, int Slen) {
    const long row = blockIdx.x;
    const float* src = scores + row * (long)Slen;
    bf16* dst = probs + row * (long)Slen;
    const int lane = threadIdx.x;

    float m = -3.0e38f;
    for (int s = lane; s < Slen; s += 32) m = fmaxf(m, src[s]);
    for (int off = 16; off; off >>= 1) m = fmaxf(m, __shfl_xor(m, off, 32));

    float sum = 0.0f;
    for (int s = lane; s < Slen; s += 32) sum += __expf(src[s] - m);
    for (int off = 16; off; off >>= 1) sum += __shfl_xor(sum, off, 32);
    const float inv = 1.0f / sum;

    for (int s = lane; s < Slen; s += 32)
        dst[s] = (bf16)(__expf(src[s] - m) * inv);
}

// head-averaged attention weights -> f32 output (B,L,Slen)
__global__ void k_headavg(const bf16* __restrict__ p, float* __restrict__ out,
                          int Slen, long n) {
    long i = (long)blockIdx.x * blockDim.x + threadIdx.x;
    if (i >= n) return;
    long s = i % Slen;
    long r = i / Slen;
    long l = r % Lm;
    long b = r / Lm;
    float acc = 0.0f;
#pragma unroll
    for (int h = 0; h < Hh; ++h)
        acc += (float)p[(((b * Hh + h) * Lm) + l) * (long)Slen + s];
    out[i] = acc * (1.0f / Hh);
}

// residual add + LayerNorm over 768; one wave per token row
__global__ __launch_bounds__(32)
void k_addln(const float* __restrict__ x, const float* __restrict__ res,
             const float* __restrict__ g, const float* __restrict__ bta,
             float* __restrict__ outf, bf16* __restrict__ outb) {
    const long row = blockIdx.x;
    const float* xr = x + row * Dm;
    const float* rr = res + row * Dm;
    const int lane = threadIdx.x;

    float vals[Dm / 32];
    float s = 0.0f;
#pragma unroll
    for (int i = 0; i < Dm / 32; ++i) {
        float v = xr[lane + 32 * i] + rr[lane + 32 * i];
        vals[i] = v; s += v;
    }
    for (int off = 16; off; off >>= 1) s += __shfl_xor(s, off, 32);
    const float mean = s * (1.0f / Dm);

    float vs = 0.0f;
#pragma unroll
    for (int i = 0; i < Dm / 32; ++i) { float d = vals[i] - mean; vs += d * d; }
    for (int off = 16; off; off >>= 1) vs += __shfl_xor(vs, off, 32);
    const float inv = rsqrtf(vs * (1.0f / Dm) + 1e-5f);

#pragma unroll
    for (int i = 0; i < Dm / 32; ++i) {
        int c = lane + 32 * i;
        float o = (vals[i] - mean) * inv * g[c] + bta[c];
        outf[row * Dm + c] = o;
        if (outb) outb[row * Dm + c] = (bf16)o;
    }
}

// ---------------------------------------------------------------------------

static inline unsigned ceilb(long n) { return (unsigned)((n + 255) / 256); }

extern "C" void kernel_launch(void* const* d_in, const int* in_sizes, int n_in,
                              void* d_out, int out_size, void* d_ws, size_t ws_size,
                              hipStream_t stream) {
    (void)in_sizes; (void)n_in; (void)out_size; (void)ws_size;

    const float* tgt      = (const float*)d_in[0];
    const float* memory   = (const float*)d_in[1];
    const float* tgt_pos  = (const float*)d_in[2];
    const float* mem_pos  = (const float*)d_in[3];
    const float* tgt_pos2 = (const float*)d_in[4];
    // weights: 5..16 sa_{qc,qp,kc,kp,v,o}_{w,b}, 17..28 ca_*, 29..32 ffn, 33..38 ln
    const float* W[14]; const float* Bb[14];
    for (int i = 0; i < 12; ++i) { W[i] = (const float*)d_in[5 + 2 * i]; Bb[i] = (const float*)d_in[6 + 2 * i]; }
    W[12] = (const float*)d_in[29]; Bb[12] = (const float*)d_in[30];   // ff1 (2048,768)
    W[13] = (const float*)d_in[31]; Bb[13] = (const float*)d_in[32];   // ff2 (768,2048)
    const float* ln_g[3] = {(const float*)d_in[33], (const float*)d_in[35], (const float*)d_in[37]};
    const float* ln_b[3] = {(const float*)d_in[34], (const float*)d_in[36], (const float*)d_in[38]};

    float* out_t   = (float*)d_out;                               // (L,B,D)
    float* out_att = out_t + (long)Lm * Bm * Dm;                  // (B,L,S)
    float* out_sat = out_att + (long)Bm * Lm * Sm;                // (B,L,L)

    // ---- workspace bump allocator (256B aligned regions) ----
    char* wsp = (char*)d_ws;
    auto alloc = [&](size_t bytes) -> void* {
        void* p = wsp; wsp += (bytes + 255) & ~(size_t)255; return p;
    };
    const long wsz[14] = {Dm*Dm,Dm*Dm,Dm*Dm,Dm*Dm,Dm*Dm,Dm*Dm,
                          Dm*Dm,Dm*Dm,Dm*Dm,Dm*Dm,Dm*Dm,Dm*Dm,
                          (long)DFFm*Dm, (long)Dm*DFFm};
    bf16* Wbf[14];
    for (int i = 0; i < 14; ++i) Wbf[i] = (bf16*)alloc(wsz[i] * 2);

    bf16* tgt_bf  = (bf16*)alloc((long)NT * Dm * 2);
    bf16* tpos_bf = (bf16*)alloc((long)NT * Dm * 2);
    bf16* mem_bf  = (bf16*)alloc((long)NS * Dm * 2);
    bf16* mpos_bf = (bf16*)alloc((long)NS * Dm * 2);
    bf16* t_bf    = (bf16*)alloc((long)NT * Dm * 2);
    bf16* aux_bf  = (bf16*)alloc((long)NT * Dm * 2);
    bf16* q_bf    = (bf16*)alloc((long)NT * 2 * Dm * 2);
    bf16* k_bf    = (bf16*)alloc((long)NS * 2 * Dm * 2);
    bf16* vt_bf   = (bf16*)alloc((long)Bm * Hh * HDm * Sm * 2);
    bf16* p_bf    = (bf16*)alloc((long)Bm * Hh * Lm * Sm * 2);
    bf16* h1_bf   = (bf16*)alloc((long)NT * DFFm * 2);

    float* bufA   = (float*)alloc((long)NS * Dm * 4);
    float* bufB   = (float*)alloc((long)NS * Dm * 4);
    float* v_f32  = (float*)alloc((long)NS * Dm * 4);
    float* scores = (float*)alloc((long)Bm * Hh * Lm * Sm * 4);
    float* o_f32  = (float*)alloc((long)NT * Dm * 4);
    float* t2_f32 = (float*)alloc((long)NT * Dm * 4);
    float* t_f32  = (float*)alloc((long)NT * Dm * 4);
    float* h1_f32 = (float*)alloc((long)NT * DFFm * 4);

    // ---- helpers ----
    auto gemm = [&](const bf16* A, long lda, long aB0, long aB1,
                    const bf16* Bw, long ldb, long bB0, long bB1,
                    const float* bias, float* C, long ldc, long cB0, long cB1,
                    int batchH, int M, int N, int K, int batch) {
        dim3 g((unsigned)(M / 128), (unsigned)(N / 64), (unsigned)batch);
        k_gemm<<<g, 128, 0, stream>>>(A, lda, aB0, aB1, Bw, ldb, bB0, bB1,
                                      bias, C, ldc, cB0, cB1, batchH, K);
    };
    auto lin = [&](const bf16* A, const bf16* Wm, const float* bias,
                   float* C, int M, int N, int K) {
        gemm(A, K, 0, 0, Wm, K, 0, 0, bias, C, N, 0, 0, 1, M, N, K, 1);
    };

    // ---- convert weights and inputs to bf16 ----
    for (int i = 0; i < 14; ++i)
        k_cvt<<<ceilb(wsz[i]), 256, 0, stream>>>(W[i], Wbf[i], wsz[i]);
    k_cvt<<<ceilb((long)NT*Dm), 256, 0, stream>>>(tgt,     tgt_bf,  (long)NT*Dm);
    k_cvt<<<ceilb((long)NT*Dm), 256, 0, stream>>>(tgt_pos, tpos_bf, (long)NT*Dm);
    k_cvt<<<ceilb((long)NS*Dm), 256, 0, stream>>>(memory,  mem_bf,  (long)NS*Dm);
    k_cvt<<<ceilb((long)NS*Dm), 256, 0, stream>>>(mem_pos, mpos_bf, (long)NS*Dm);

    // ================= self attention =================
    lin(tpos_bf, Wbf[0], Bb[0], bufA, NT, Dm, Dm);                 // sa_qc(tgt_pos)
    lin(tgt_bf,  Wbf[1], Bb[1], bufB, NT, Dm, Dm);                 // sa_qp(tgt)
    k_fuse<<<ceilb((long)NT*2*Dm), 256, 0, stream>>>(bufA, bufB, q_bf, SCALE, (long)NT*2*Dm);
    lin(tpos_bf, Wbf[2], Bb[2], bufA, NT, Dm, Dm);                 // sa_kc(tgt_pos)
    lin(tgt_bf,  Wbf[3], Bb[3], bufB, NT, Dm, Dm);                 // sa_kp(tgt)
    k_fuse<<<ceilb((long)NT*2*Dm), 256, 0, stream>>>(bufA, bufB, k_bf, 1.0f, (long)NT*2*Dm);
    lin(tgt_bf,  Wbf[4], Bb[4], v_f32, NT, Dm, Dm);                // sa_v(tgt)
    k_vt<<<ceilb((long)Bm*Hh*HDm*Lm), 256, 0, stream>>>(v_f32, vt_bf, Lm, (long)Bm*Hh*HDm*Lm);

    // scores[b,h,l,s] = q . k  (K = 2*HD = 256), 48 batches
    gemm(q_bf, (long)Bm*2*Dm, 2*Dm, 2*HDm,
         k_bf, (long)Bm*2*Dm, 2*Dm, 2*HDm,
         nullptr, scores, Lm, (long)Hh*Lm*Lm, (long)Lm*Lm,
         Hh, Lm, Lm, 2*HDm, Bm*Hh);
    k_softmax<<<Bm*Hh*Lm, 32, 0, stream>>>(scores, p_bf, Lm);
    k_headavg<<<ceilb((long)Bm*Lm*Lm), 256, 0, stream>>>(p_bf, out_sat, Lm, (long)Bm*Lm*Lm);

    // O = P @ V  (N = HD = 128, K = L)
    gemm(p_bf, Lm, (long)Hh*Lm*Lm, (long)Lm*Lm,
         vt_bf, Lm, (long)Hh*HDm*Lm, (long)HDm*Lm,
         nullptr, o_f32, (long)Bm*Dm, Dm, HDm,
         Hh, Lm, HDm, Lm, Bm*Hh);
    k_cvt<<<ceilb((long)NT*Dm), 256, 0, stream>>>(o_f32, aux_bf, (long)NT*Dm);
    lin(aux_bf, Wbf[5], Bb[5], t2_f32, NT, Dm, Dm);                // sa_o
    k_addln<<<NT, 32, 0, stream>>>(t2_f32, tgt, ln_g[0], ln_b[0], t_f32, t_bf);

    // ================= cross attention =================
    lin(tpos_bf, Wbf[6], Bb[6], bufA, NT, Dm, Dm);                 // ca_qc(tgt_pos)
    k_addcvt<<<ceilb((long)NT*Dm), 256, 0, stream>>>(t_f32, tgt_pos2, aux_bf, (long)NT*Dm);
    lin(aux_bf, Wbf[7], Bb[7], bufB, NT, Dm, Dm);                  // ca_qp(t + tgt_pos2)
    k_fuse<<<ceilb((long)NT*2*Dm), 256, 0, stream>>>(bufA, bufB, q_bf, SCALE, (long)NT*2*Dm);
    lin(mem_bf,  Wbf[8], Bb[8], bufA, NS, Dm, Dm);                 // ca_kc(memory)
    lin(mpos_bf, Wbf[9], Bb[9], bufB, NS, Dm, Dm);                 // ca_kp(memory_pos)
    k_fuse<<<ceilb((long)NS*2*Dm), 256, 0, stream>>>(bufA, bufB, k_bf, 1.0f, (long)NS*2*Dm);
    lin(mpos_bf, Wbf[10], Bb[10], v_f32, NS, Dm, Dm);              // ca_v(memory_pos)
    k_vt<<<ceilb((long)Bm*Hh*HDm*Sm), 256, 0, stream>>>(v_f32, vt_bf, Sm, (long)Bm*Hh*HDm*Sm);

    gemm(q_bf, (long)Bm*2*Dm, 2*Dm, 2*HDm,
         k_bf, (long)Bm*2*Dm, 2*Dm, 2*HDm,
         nullptr, scores, Sm, (long)Hh*Lm*Sm, (long)Lm*Sm,
         Hh, Lm, Sm, 2*HDm, Bm*Hh);
    k_softmax<<<Bm*Hh*Lm, 32, 0, stream>>>(scores, p_bf, Sm);
    k_headavg<<<ceilb((long)Bm*Lm*Sm), 256, 0, stream>>>(p_bf, out_att, Sm, (long)Bm*Lm*Sm);

    gemm(p_bf, Sm, (long)Hh*Lm*Sm, (long)Lm*Sm,
         vt_bf, Sm, (long)Hh*HDm*Sm, (long)HDm*Sm,
         nullptr, o_f32, (long)Bm*Dm, Dm, HDm,
         Hh, Lm, HDm, Sm, Bm*Hh);
    k_cvt<<<ceilb((long)NT*Dm), 256, 0, stream>>>(o_f32, aux_bf, (long)NT*Dm);
    lin(aux_bf, Wbf[11], Bb[11], t2_f32, NT, Dm, Dm);              // ca_o
    k_addln<<<NT, 32, 0, stream>>>(t2_f32, t_f32, ln_g[1], ln_b[1], t_f32, t_bf);

    // ================= FFN + post-norm =================
    lin(t_bf, Wbf[12], Bb[12], h1_f32, NT, DFFm, Dm);              // ff1
    k_relu<<<ceilb((long)NT*DFFm), 256, 0, stream>>>(h1_f32, h1_bf, (long)NT*DFFm);
    lin(h1_bf, Wbf[13], Bb[13], t2_f32, NT, Dm, DFFm);             // ff2
    k_addln<<<NT, 32, 0, stream>>>(t2_f32, t_f32, ln_g[2], ln_b[2], out_t, (bf16*)nullptr);
}
